// CC_module_6365141533261
// MI455X (gfx1250) — matrix operand order repaired and checked by reading, not compiled, not run on hardware
//
#include <hip/hip_runtime.h>
#include <hip/hip_bf16.h>

typedef __attribute__((ext_vector_type(16))) _Float16 v16h;
typedef __attribute__((ext_vector_type(8)))  _Float16 v8h;
typedef __attribute__((ext_vector_type(8)))  float    v8f;

#define B_    4
#define C_    256
#define H_    128
#define W_    128
#define HW_   (H_*W_)
#define CHW_  (C_*HW_)
#define CQK_  32
#define INH_  129       // 129 grid points per deformed line
#define NQK_  320       // 32 (q) + 32 (k) + 256 (v) fused output channels
#define LINES_ 512      // B*W == B*H

union AF { v16h v; _Float16 h[16]; };

__device__ __forceinline__ v8f wmma_f16(v16h a, v16h b, v8f c) {
  // D = A(16x32 f16) * B(32x16 f16) + C(16x16 f32)
  return __builtin_amdgcn_wmma_f32_16x16x32_f16(false, a, false, b, (short)0, c,
                                                false, false);
}

// Fragment-order index for a B operand element (k, n) of a GEMM whose N spans
// `ntiles` 16-wide tiles. Layout: [kt][nt][lane(32)][half(16)] so each lane's
// 16 halves are 32 contiguous bytes (two b128 loads).
__device__ __forceinline__ int fragIdx(int k, int n, int ntiles) {
  const int kt = k >> 5, kr = k & 31;
  const int l  = (((kr >> 3) & 1) ? 16 : 0) + (n & 15);
  return (((kt * ntiles) + (n >> 4)) * 32 + l) * 16 + ((kr & 7) + ((kr >> 4) << 3));
}

// ---------------------------------------------------------------------------
// K0: offset pooling -> dX/dY lines, linspace, fused f16 weights (fragment
//     order), bias, regloss
// ---------------------------------------------------------------------------
__global__ void k0_prep(const float* __restrict__ off,
                        const float* __restrict__ Wq, const float* __restrict__ bq,
                        const float* __restrict__ Wk, const float* __restrict__ bk,
                        const float* __restrict__ Wv, const float* __restrict__ bv,
                        float* __restrict__ dX, float* __restrict__ dY,
                        float* __restrict__ lin,
                        _Float16* __restrict__ WallT, float* __restrict__ biasAll,
                        float* __restrict__ out_reg) {
  const int tid = threadIdx.x;
  // 3x3 avg pool of offsets[0] (2x131x131 -> 2x129x129); keep col 64 / row 64
  for (int t = tid; t < 2 * INH_; t += blockDim.x) {
    int r = t >> 1, ch = t & 1;
    float s = 0.f;
    for (int i = 0; i < 3; ++i)
      for (int j = 0; j < 3; ++j)
        s += off[ch * 17161 + (r + i) * 131 + (64 + j)];
    dX[r * 2 + ch] = s * (1.0f / 9.0f);
  }
  for (int t = tid; t < 2 * INH_; t += blockDim.x) {
    int c = t >> 1, ch = t & 1;
    float s = 0.f;
    for (int i = 0; i < 3; ++i)
      for (int j = 0; j < 3; ++j)
        s += off[ch * 17161 + (64 + i) * 131 + (c + j)];
    dY[c * 2 + ch] = s * (1.0f / 9.0f);
  }
  for (int t = tid; t < INH_; t += blockDim.x)
    lin[t] = (float)t * (2.0f / 128.0f) - 1.0f;

  // fused weight in B-fragment order: GEMM K=256 (8 k-tiles), N=320 (20 tiles)
  for (int idx = tid; idx < 256 * NQK_; idx += blockDim.x) {
    int k = idx / NQK_, n = idx % NQK_;
    float v;
    if (n < 32)       v = Wq[n * 256 + k];
    else if (n < 64)  v = Wk[(n - 32) * 256 + k];
    else              v = Wv[(n - 64) * 256 + k];
    WallT[fragIdx(k, n, 20)] = (_Float16)v;
  }
  for (int t = tid; t < NQK_; t += blockDim.x)
    biasAll[t] = (t < 32) ? bq[t] : (t < 64) ? bk[t - 32] : bv[t - 64];

  // regloss = mean((off0-off1)^2) + mean((off[:,r]-off[:,r+1])^2)
  float local = 0.f;
  for (int p = tid; p < 17161; p += blockDim.x) {
    float d = off[p] - off[17161 + p];
    local += d * d * (1.0f / 17161.0f);
  }
  for (int q = tid; q < 34060; q += blockDim.x) {
    int ch = q / 17030, rem = q % 17030, r = rem / 131, c = rem % 131;
    float d = off[ch * 17161 + r * 131 + c] - off[ch * 17161 + (r + 1) * 131 + c];
    local += d * d * (1.0f / 34060.0f);
  }
  __shared__ float red[256];
  red[tid] = local;
  __syncthreads();
  for (int s = 128; s > 0; s >>= 1) {
    if (tid < s) red[tid] += red[tid + s];
    __syncthreads();
  }
  if (tid == 0) *out_reg = red[0];
}

// ---------------------------------------------------------------------------
// K1: fused projection GEMM  out[(b,h,w), n] = x[(b,h,w), c] * W[c][n] + bias
//     M=65536  N=320  K=256. A staged through LDS (coalesced fp32 loads ->
//     f16 tile -> ds_load_b128 fragments); B read as contiguous b128 frags.
// ---------------------------------------------------------------------------
#define LDS_STRIDE 40   // 32 halves + 8 pad: keeps 16B alignment, spreads banks

__global__ void k1_proj(const float* __restrict__ x,
                        const _Float16* __restrict__ WallT,
                        const float* __restrict__ biasAll,
                        _Float16* __restrict__ qH, _Float16* __restrict__ kH,
                        _Float16* __restrict__ vH,
                        _Float16* __restrict__ qW, _Float16* __restrict__ kW,
                        _Float16* __restrict__ vW) {
  __shared__ _Float16 ldsA[128 * LDS_STRIDE];
  const int tid  = threadIdx.x;
  const int wave = tid >> 5, lane = tid & 31;
  const int mrow = lane & 15;
  const int kb   = (lane < 16) ? 0 : 8;    // A-frag K sub-chunk base
  const int n_l  = lane & 15;
  const int m0   = blockIdx.x * 128;       // block-row base (never crosses b)
  const int b0   = m0 >> 14, hw0 = m0 & 16383;
  const float* xb = x + b0 * CHW_ + hw0;

  // Stage all 8 K-tiles of A through LDS; keep fragments in registers.
  AF a[8];
  for (int kt = 0; kt < 8; ++kt) {
    for (int idx = tid; idx < 4096; idx += 256) {
      const int k = idx >> 7, m = idx & 127;        // consecutive tid -> m
      ldsA[m * LDS_STRIDE + k] = (_Float16)xb[(kt * 32 + k) * HW_ + m];
    }
    __syncthreads();
    const int row = wave * 16 + mrow;
    a[kt].v = {};
    *(v8h*)&a[kt].h[0] = *(const v8h*)&ldsA[row * LDS_STRIDE + kb];
    *(v8h*)&a[kt].h[8] = *(const v8h*)&ldsA[row * LDS_STRIDE + 16 + kb];
    __syncthreads();
  }

  for (int nt = 0; nt < 20; ++nt) {
    v8f acc = {};
    for (int kt = 0; kt < 8; ++kt) {
      const _Float16* bp = WallT + ((kt * 20 + nt) * 32 + lane) * 16;
      AF bf;
      *(v8h*)&bf.h[0] = *(const v8h*)(bp);
      *(v8h*)&bf.h[8] = *(const v8h*)(bp + 8);
      acc = wmma_f16(a[kt].v, bf.v, acc);
    }
    const int n = nt * 16 + n_l;
    const float bias = biasAll[n];
#pragma unroll
    for (int r = 0; r < 8; ++r) {
      const int mo  = blockIdx.x * 128 + wave * 16 + r + ((lane >= 16) ? 8 : 0);
      const int bo  = mo >> 14, hwo = mo & 16383;
      const int ho  = hwo >> 7, wo = hwo & 127;
      const _Float16 val = (_Float16)(acc[r] + bias);
      if (n < 32) {
        qH[((bo * W_ + wo) * H_ + ho) * CQK_ + n] = val;   // A-operand: row major
        qW[((bo * H_ + ho) * W_ + wo) * CQK_ + n] = val;
      } else if (n < 64) {
        const int o = n - 32;                               // B-operand: frag order
        kH[(bo * W_ + wo) * 4096 + fragIdx(o, ho, 8)] = val;
        kW[(bo * H_ + ho) * 4096 + fragIdx(o, wo, 8)] = val;
      } else {
        const int o = n - 64;                               // A-operand: row major
        vH[((bo * W_ + wo) * C_ + o) * H_ + ho] = val;
        vW[((bo * H_ + ho) * C_ + o) * W_ + wo] = val;
      }
    }
  }
}

// ---------------------------------------------------------------------------
// K2a: deformable bilinear sampling of batch 0 along offset-deformed lines.
//      Reads the (still original) W-path layouts, writes temp buffers.
// ---------------------------------------------------------------------------
__device__ __forceinline__ float fetch_proj(const _Float16* qW, const _Float16* kW,
                                            const _Float16* vW, int c, int y, int x) {
  if (c < 32)      return (float)qW[(y * W_ + x) * CQK_ + c];
  else if (c < 64) return (float)kW[y * 4096 + fragIdx(c - 32, x, 8)];
  else             return (float)vW[(y * C_ + (c - 64)) * W_ + x];
}

__global__ void k2a_sample(const _Float16* __restrict__ qW,
                           const _Float16* __restrict__ kW,
                           const _Float16* __restrict__ vW,
                           const float* __restrict__ dX, const float* __restrict__ dY,
                           const float* __restrict__ lin,
                           _Float16* __restrict__ tmpH, _Float16* __restrict__ tmpW) {
  const int i = blockIdx.x, j = blockIdx.y, path = blockIdx.z;
  float gx, gy;
  if (path == 0) {  // column lines: PH[i][j+1]
    gx = lin[i] + dX[(j + 1) * 2 + 0];
    gy = lin[j + 1] + dX[(j + 1) * 2 + 1];
  } else {          // row lines: PW[i][j+1]
    gx = lin[j + 1] + dY[(j + 1) * 2 + 0];
    gy = lin[i] + dY[(j + 1) * 2 + 1];
  }
  const float fx = (gx + 1.0f) * 0.5f * 127.0f;
  const float fy = (gy + 1.0f) * 0.5f * 127.0f;
  const int x0 = (int)floorf(fx), y0 = (int)floorf(fy);
  const float wx1 = fx - (float)x0, wy1 = fy - (float)y0;
  const float wgt[4] = {(1.f - wx1) * (1.f - wy1), wx1 * (1.f - wy1),
                        (1.f - wx1) * wy1,         wx1 * wy1};
  const int xs[4] = {x0, x0 + 1, x0,     x0 + 1};
  const int ys[4] = {y0, y0,     y0 + 1, y0 + 1};
  _Float16* dst = (path == 0 ? tmpH : tmpW) + (i * 128 + j) * NQK_;
  for (int c = threadIdx.x; c < NQK_; c += blockDim.x) {
    float acc = 0.f;
#pragma unroll
    for (int t = 0; t < 4; ++t) {
      const int xi = xs[t], yi = ys[t];
      if (xi >= 0 && xi < 128 && yi >= 0 && yi < 128)
        acc += wgt[t] * fetch_proj(qW, kW, vW, c, yi, xi);
    }
    dst[c] = (_Float16)acc;
  }
}

// K2b: scatter sampled values over the b=0 slices of all six tensors
__global__ void k2b_scatter(const _Float16* __restrict__ tmpH,
                            const _Float16* __restrict__ tmpW,
                            _Float16* __restrict__ qH, _Float16* __restrict__ kH,
                            _Float16* __restrict__ vH,
                            _Float16* __restrict__ qW, _Float16* __restrict__ kW,
                            _Float16* __restrict__ vW) {
  const int i = blockIdx.x, j = blockIdx.y, path = blockIdx.z;
  const _Float16* src = (path == 0 ? tmpH : tmpW) + (i * 128 + j) * NQK_;
  for (int c = threadIdx.x; c < NQK_; c += blockDim.x) {
    const _Float16 v = src[c];
    if (path == 0) {
      if (c < 32)      qH[(i * H_ + j) * CQK_ + c] = v;
      else if (c < 64) kH[i * 4096 + fragIdx(c - 32, j, 8)] = v;
      else             vH[(i * C_ + (c - 64)) * H_ + j] = v;
    } else {
      if (c < 32)      qW[(i * W_ + j) * CQK_ + c] = v;
      else if (c < 64) kW[i * 4096 + fragIdx(c - 32, j, 8)] = v;
      else             vW[(i * C_ + (c - 64)) * W_ + j] = v;
    }
  }
}

// ---------------------------------------------------------------------------
// K3: attention scores. 512 batched 128x128 GEMMs with K=32 (one WMMA/tile),
//     both operands consumed as two b128 loads per lane.
// ---------------------------------------------------------------------------
__global__ void k3_scores(const _Float16* __restrict__ qH,
                          const _Float16* __restrict__ kH,
                          const _Float16* __restrict__ qW,
                          const _Float16* __restrict__ kW,
                          float* __restrict__ att) {
  const int batch = blockIdx.x, path = blockIdx.y;
  const int wave = threadIdx.x >> 5, lane = threadIdx.x & 31;
  const _Float16* q = (path == 0 ? qH : qW) + batch * H_ * CQK_;
  const _Float16* k = (path == 0 ? kH : kW) + batch * 4096;
  const int mrow = lane & 15;
  const int kb   = (lane < 16) ? 0 : 8;
  const int n_l  = lane & 15;
  const int bb   = batch >> 7, line = batch & 127;

  for (int t = wave; t < 64; t += 8) {
    const int tm = t >> 3, tn = t & 7;
    AF a;
    *(v8h*)&a.h[0] = *(const v8h*)(q + (tm * 16 + mrow) * CQK_ + kb);
    *(v8h*)&a.h[8] = *(const v8h*)(q + (tm * 16 + mrow) * CQK_ + 16 + kb);
    const _Float16* bp = k + (tn * 32 + lane) * 16;
    AF bf;
    *(v8h*)&bf.h[0] = *(const v8h*)(bp);
    *(v8h*)&bf.h[8] = *(const v8h*)(bp + 8);
    v8f acc = {};
    acc = wmma_f16(a.v, bf.v, acc);
#pragma unroll
    for (int r = 0; r < 8; ++r) {
      const int M = tm * 16 + r + ((lane >= 16) ? 8 : 0);
      const int N = tn * 16 + n_l;
      float val = acc[r];
      if (path == 0) {
        if (N == M) val = -__builtin_inff();
        att[((bb * H_ + M) * W_ + line) * 256 + N] = val;       // [b][h=M][w][j]
      } else {
        att[((bb * H_ + line) * W_ + M) * 256 + 128 + N] = val; // [b][h][w=M][H+j]
      }
    }
  }
}

// ---------------------------------------------------------------------------
// K4: joint softmax over 256 logits per (b,h,w); one wave per row.
//     Writes f16 probabilities directly in B-fragment order for the AV GEMM.
// ---------------------------------------------------------------------------
__global__ void k4_softmax(const float* __restrict__ att,
                           _Float16* __restrict__ ptH, _Float16* __restrict__ ptW) {
  const int wave = threadIdx.x >> 5, lane = threadIdx.x & 31;
  const int rid = blockIdx.x * 8 + wave;
  const float* row = att + (size_t)rid * 256;
  float v[8];
  float mx = -__builtin_inff();
#pragma unroll
  for (int i = 0; i < 8; ++i) { v[i] = row[lane + 32 * i]; mx = fmaxf(mx, v[i]); }
  for (int o = 16; o >= 1; o >>= 1) mx = fmaxf(mx, __shfl_xor(mx, o, 32));
  float sum = 0.f;
#pragma unroll
  for (int i = 0; i < 8; ++i) { v[i] = __expf(v[i] - mx); sum += v[i]; }
  for (int o = 16; o >= 1; o >>= 1) sum += __shfl_xor(sum, o, 32);
  const float inv = 1.0f / sum;
  const int bb = rid >> 14, rem = rid & 16383, h = rem >> 7, w = rem & 127;
#pragma unroll
  for (int i = 0; i < 8; ++i) {
    const int t = lane + 32 * i;
    const _Float16 p = (_Float16)(v[i] * inv);
    if (t < 128) ptH[(bb * W_ + w) * 16384 + fragIdx(t, h, 8)] = p;        // k=j, n=h
    else         ptW[(bb * H_ + h) * 16384 + fragIdx(t - 128, w, 8)] = p;  // k=j, n=w
  }
}

// ---------------------------------------------------------------------------
// K5: Att x V. 512 batched 256x128 GEMMs with K=128 (4 WMMA steps); both
//     operands as b128 fragment loads. path 0: out = x + g*outH;
//     path 1: out += g*outW.
// ---------------------------------------------------------------------------
__global__ void k5_av(const _Float16* __restrict__ vH, const _Float16* __restrict__ vW,
                      const _Float16* __restrict__ ptH, const _Float16* __restrict__ ptW,
                      const float* __restrict__ x, const float* __restrict__ gamma,
                      float* __restrict__ out, int path) {
  const int batch = blockIdx.x, tm = blockIdx.y;
  const int wave = threadIdx.x >> 5, lane = threadIdx.x & 31;
  const int tn = wave;
  const _Float16* vp = (path == 0 ? vH : vW) + (size_t)batch * C_ * H_ + tm * 16 * H_;
  const _Float16* pp = (path == 0 ? ptH : ptW) + (size_t)batch * 16384;
  const int mrow = lane & 15;
  const int kb   = (lane < 16) ? 0 : 8;
  const int n_l  = lane & 15;

  v8f acc = {};
  for (int kt = 0; kt < 4; ++kt) {
    const int kk = kt * 32;
    const _Float16* bp = pp + ((kt * 8 + tn) * 32 + lane) * 16;
    if (kt < 3) __builtin_prefetch(pp + (((kt + 1) * 8 + tn) * 32 + lane) * 16, 0, 1);
    AF a;
    *(v8h*)&a.h[0] = *(const v8h*)(vp + mrow * H_ + kk + kb);
    *(v8h*)&a.h[8] = *(const v8h*)(vp + mrow * H_ + kk + 16 + kb);
    AF bf;
    *(v8h*)&bf.h[0] = *(const v8h*)(bp);
    *(v8h*)&bf.h[8] = *(const v8h*)(bp + 8);
    acc = wmma_f16(a.v, bf.v, acc);
  }

  const float g = gamma[0];
  const int bb = batch >> 7, line = batch & 127;
#pragma unroll
  for (int r = 0; r < 8; ++r) {
    const int c = tm * 16 + r + ((lane >= 16) ? 8 : 0);
    const int n = tn * 16 + n_l;
    if (path == 0) {
      const int idx = bb * CHW_ + c * HW_ + n * W_ + line;  // out[b][c][h=n][w=line]
      out[idx] = x[idx] + g * acc[r];
    } else {
      const int idx = bb * CHW_ + c * HW_ + line * W_ + n;  // out[b][c][h=line][w=n]
      out[idx] += g * acc[r];
    }
  }
}

// ---------------------------------------------------------------------------
extern "C" void kernel_launch(void* const* d_in, const int* in_sizes, int n_in,
                              void* d_out, int out_size, void* d_ws, size_t ws_size,
                              hipStream_t stream) {
  (void)in_sizes; (void)n_in; (void)ws_size;
  const float* x   = (const float*)d_in[0];
  const float* Wq  = (const float*)d_in[1];
  const float* bq  = (const float*)d_in[2];
  const float* Wk  = (const float*)d_in[3];
  const float* bk  = (const float*)d_in[4];
  const float* Wv  = (const float*)d_in[5];
  const float* bv  = (const float*)d_in[6];
  const float* gm  = (const float*)d_in[7];
  const float* off = (const float*)d_in[8];
  float* out = (float*)d_out;

  char* p = (char*)d_ws;
  auto carve = [&](size_t bytes) -> void* {
    void* r = (void*)p;
    p += (bytes + 255) & ~(size_t)255;
    return r;
  };
  float*    dX      = (float*)carve(2 * INH_ * sizeof(float));
  float*    dY      = (float*)carve(2 * INH_ * sizeof(float));
  float*    lin     = (float*)carve(INH_ * sizeof(float));
  _Float16* WallT   = (_Float16*)carve((size_t)256 * NQK_ * 2);
  float*    biasAll = (float*)carve(NQK_ * sizeof(float));
  _Float16* qH = (_Float16*)carve((size_t)LINES_ * H_ * CQK_ * 2);   // [bW][h][cq]
  _Float16* kH = (_Float16*)carve((size_t)LINES_ * CQK_ * H_ * 2);   // [bW] frag order
  _Float16* qW = (_Float16*)carve((size_t)LINES_ * W_ * CQK_ * 2);   // [bH][w][cq]
  _Float16* kW = (_Float16*)carve((size_t)LINES_ * CQK_ * W_ * 2);   // [bH] frag order
  _Float16* vH = (_Float16*)carve((size_t)LINES_ * C_ * H_ * 2);     // [bW][c][h]
  _Float16* vW = (_Float16*)carve((size_t)LINES_ * C_ * W_ * 2);     // [bH][c][w]
  _Float16* tmpH = (_Float16*)carve((size_t)128 * 128 * NQK_ * 2);
  _Float16* tmpW = (_Float16*)carve((size_t)128 * 128 * NQK_ * 2);
  float*    att  = (float*)carve((size_t)B_ * H_ * W_ * 256 * sizeof(float));
  _Float16* ptH  = (_Float16*)carve((size_t)LINES_ * H_ * H_ * 2);   // frag order
  _Float16* ptW  = (_Float16*)carve((size_t)LINES_ * W_ * W_ * 2);   // frag order

  k0_prep<<<1, 256, 0, stream>>>(off, Wq, bq, Wk, bk, Wv, bv, dX, dY, lin,
                                 WallT, biasAll, out + (out_size - 1));
  k1_proj<<<512, 256, 0, stream>>>(x, WallT, biasAll, qH, kH, vH, qW, kW, vW);
  k2a_sample<<<dim3(128, 128, 2), 64, 0, stream>>>(qW, kW, vW, dX, dY, lin, tmpH, tmpW);
  k2b_scatter<<<dim3(128, 128, 2), 64, 0, stream>>>(tmpH, tmpW, qH, kH, vH, qW, kW, vW);
  k3_scores<<<dim3(512, 2), 256, 0, stream>>>(qH, kH, qW, kW, att);
  k4_softmax<<<8192, 256, 0, stream>>>(att, ptH, ptW);
  k5_av<<<dim3(512, 16), 256, 0, stream>>>(vH, vW, ptH, ptW, x, gm, out, 0);
  k5_av<<<dim3(512, 16), 256, 0, stream>>>(vH, vW, ptH, ptW, x, gm, out, 1);
}
